// MaskingNet_14654428414190
// MI455X (gfx1250) — matrix operand order repaired and checked by reading, compiled, not run on hardware
//
#include <hip/hip_runtime.h>

#define DD 4096
#define NW (DD*DD)
#define KW (NW/2)
#define KB (DD/2)

typedef __attribute__((ext_vector_type(16))) __bf16 v16bf;
typedef __attribute__((ext_vector_type(8)))  float  v8f;

__device__ inline unsigned short f2bf(float f) {
    unsigned u = __float_as_uint(f);
    unsigned r = u + 0x7FFFu + ((u >> 16) & 1u);   // round-to-nearest-even
    return (unsigned short)(r >> 16);
}

// ---- fp32 -> bf16 conversion of activations ----
__global__ void k_convert(const float* __restrict__ x, unsigned short* __restrict__ out, int n) {
    int i = blockIdx.x * blockDim.x + threadIdx.x;
    int stride = gridDim.x * blockDim.x;
    for (; i < n; i += stride) out[i] = f2bf(x[i]);
}

// ---- radix-select state init ----
__global__ void k_sel_init(unsigned* hist, uint2* sel, unsigned k) {
    hist[threadIdx.x] = 0u;
    if (threadIdx.x == 0) { sel->x = 0u; sel->y = k; }   // x = prefix, y = rank remaining (1-indexed)
}

// ---- one radix round: histogram of byte (bits >> shift) among elements matching prefix ----
__global__ void k_hist(const unsigned* __restrict__ mbits, int n, unsigned shift,
                       const uint2* __restrict__ sel, unsigned* __restrict__ hist) {
    __shared__ unsigned sh[256];
    sh[threadIdx.x] = 0u;
    __syncthreads();
    unsigned prefix = sel->x;
    int i = blockIdx.x * blockDim.x + threadIdx.x;
    int stride = gridDim.x * blockDim.x;
    for (; i < n; i += stride) {
        unsigned b = mbits[i];
        unsigned hi   = (shift == 24u) ? 0u : (b >> (shift + 8u));
        unsigned want = (shift == 24u) ? 0u : prefix;
        if (hi == want) atomicAdd(&sh[(b >> shift) & 0xFFu], 1u);
    }
    __syncthreads();
    atomicAdd(&hist[threadIdx.x], sh[threadIdx.x]);
}

// ---- pick bucket containing the r-th smallest, extend prefix, reset hist ----
__global__ void k_scan(unsigned* hist, uint2* sel) {
    if (threadIdx.x == 0) {
        unsigned r = sel->y;
        unsigned cum = 0, b = 255, rn = r;
        for (unsigned i = 0; i < 256; ++i) {
            unsigned c = hist[i];
            if (cum + c >= r) { b = i; rn = r - cum; break; }
            cum += c;
        }
        sel->x = (sel->x << 8) | b;
        sel->y = rn;
    }
    __syncthreads();
    hist[threadIdx.x] = 0u;
}

// ---- fused mask + fp32->bf16 weight conversion ----
__global__ void k_mask_w(const float* __restrict__ W, const unsigned* __restrict__ mbits,
                         const uint2* __restrict__ sel, unsigned short* __restrict__ wm, int n) {
    unsigned t = sel->x;  // exact bit pattern of k-th smallest mask value
    int i = blockIdx.x * blockDim.x + threadIdx.x;
    int stride = gridDim.x * blockDim.x;
    for (; i < n; i += stride)
        wm[i] = (mbits[i] <= t) ? (unsigned short)0 : f2bf(W[i]);
}

// ---- exact bias masking (rank with stable tie-break, matches top_k semantics) ----
__global__ void k_mask_bias(const float* __restrict__ b, const float* __restrict__ mb,
                            float* __restrict__ out) {
    __shared__ float v[DD];
    for (int i = threadIdx.x; i < DD; i += blockDim.x) v[i] = mb[i];
    __syncthreads();
    for (int i = threadIdx.x; i < DD; i += blockDim.x) {
        float vi = v[i];
        int rank = 0;
        for (int j = 0; j < DD; ++j) {
            float vj = v[j];
            rank += (vj < vi) || (vj == vi && j < i);
        }
        out[i] = (rank < KB) ? 0.0f : b[i];
    }
}

// ---- WMMA GEMM: out[m][n] = sum_k A[m][k]*Wm[n][k] + bias[n], optional relu ----
// Block tile 128x128, 8 waves, each wave 64(M)x32(N); K-step 32.
// Double-buffered LDS; tiles staged with CDNA5 async global->LDS copies (ASYNCcnt),
// prefetch distance 1 tile: s_wait_asynccnt 4 only blocks on the *previous* tile.
#define BM 128
#define BN 128
#define BK 32
#define LDT 40   // padded LDS row stride (ushorts): 80B rows, bank-conflict free, 16B aligned

__device__ __forceinline__ void stage_issue(unsigned lA, unsigned lB,
                                            const unsigned short* gA, const unsigned short* gB) {
    asm volatile("global_load_async_to_lds_b128 %0, %1, off"           :: "v"(lA), "v"(gA) : "memory");
    asm volatile("global_load_async_to_lds_b128 %0, %1, off offset:16" :: "v"(lA), "v"(gA) : "memory");
    asm volatile("global_load_async_to_lds_b128 %0, %1, off"           :: "v"(lB), "v"(gB) : "memory");
    asm volatile("global_load_async_to_lds_b128 %0, %1, off offset:16" :: "v"(lB), "v"(gB) : "memory");
}

__device__ __forceinline__ void compute_tile(const unsigned short* Asb, const unsigned short* Bsb,
                                             v8f (&acc)[4][2], int wmi, int wni, int lr, int lh) {
    union Frag { unsigned u[8]; v16bf v; };
    Frag af[4], bf[2];
    // A fragment: 16-bit A layout — VGPR v holds K = (v<4?0:16) + (half?8:0) + 2*(v%4), pairs
    #pragma unroll
    for (int mt = 0; mt < 4; ++mt) {
        int m = wmi * 64 + mt * 16 + lr;
        #pragma unroll
        for (int v = 0; v < 8; ++v) {
            int k = ((v < 4) ? 0 : 16) + (lh ? 8 : 0) + 2 * (v & 3);
            af[mt].u[v] = *(const unsigned*)&Asb[m * LDT + k];
        }
    }
    // B fragment: same K layout as A — VGPR v holds K = (v<4?0:16) + (half?8:0) + 2*(v%4), pairs
    #pragma unroll
    for (int nt = 0; nt < 2; ++nt) {
        int n = wni * 32 + nt * 16 + lr;
        #pragma unroll
        for (int v = 0; v < 8; ++v) {
            int k = ((v < 4) ? 0 : 16) + (lh ? 8 : 0) + 2 * (v & 3);
            bf[nt].u[v] = *(const unsigned*)&Bsb[n * LDT + k];
        }
    }
    #pragma unroll
    for (int mt = 0; mt < 4; ++mt)
        #pragma unroll
        for (int nt = 0; nt < 2; ++nt)
            acc[mt][nt] = __builtin_amdgcn_wmma_f32_16x16x32_bf16(
                false, af[mt].v, false, bf[nt].v,
                (short)0, acc[mt][nt], false, false);
}

__global__ __launch_bounds__(256)
void k_gemm(const unsigned short* __restrict__ A,   // [4096][4096] bf16 activations (row-major)
            const unsigned short* __restrict__ Wm,  // [4096][4096] bf16 masked weights (row-major)
            const float* __restrict__ bias,
            unsigned short* __restrict__ outb, float* __restrict__ outf,
            int relu, int storef32) {
    __shared__ unsigned short As[2][BM * LDT];
    __shared__ unsigned short Bs[2][BN * LDT];

    const int tid = threadIdx.x;
    const int m0 = blockIdx.y * BM;
    const int n0 = blockIdx.x * BN;
    const int wave = tid >> 5;
    const int lane = tid & 31;
    const int wmi = wave & 1;    // wave M index (0..1) -> 64 rows each
    const int wni = wave >> 1;   // wave N index (0..3) -> 32 cols each
    const int lr = lane & 15;
    const int lh = lane >> 4;

    v8f z = {};
    v8f acc[4][2];
    #pragma unroll
    for (int i = 0; i < 4; ++i) { acc[i][0] = z; acc[i][1] = z; }

    // staging assignment: each thread owns one 16-elem (32B) chunk of A and of B
    const int srow = tid >> 1;          // 0..127
    const int scol = (tid & 1) * 16;    // 0 / 16

    const unsigned lA0 = (unsigned)(size_t)&As[0][srow * LDT + scol];
    const unsigned lA1 = (unsigned)(size_t)&As[1][srow * LDT + scol];
    const unsigned lB0 = (unsigned)(size_t)&Bs[0][srow * LDT + scol];
    const unsigned lB1 = (unsigned)(size_t)&Bs[1][srow * LDT + scol];
    const unsigned short* gA = A  + (size_t)(m0 + srow) * DD + scol;
    const unsigned short* gB = Wm + (size_t)(n0 + srow) * DD + scol;

    // prologue: prefetch tile 0 into buffer 0
    stage_issue(lA0, lB0, gA, gB); gA += BK; gB += BK;

    for (int k0 = 0; k0 < DD; k0 += 2 * BK) {
        // ---- stage 0: prefetch into buf1, compute buf0 ----
        __syncthreads();                                  // all waves done reading buf1
        stage_issue(lA1, lB1, gA, gB); gA += BK; gB += BK;
        asm volatile("s_wait_asynccnt 0x4" ::: "memory"); // previous tile (buf0) landed
        __syncthreads();                                  // visible to all waves
        compute_tile(As[0], Bs[0], acc, wmi, wni, lr, lh);

        // ---- stage 1: prefetch into buf0, compute buf1 ----
        __syncthreads();                                  // all waves done reading buf0
        // (final iteration over-fetches one tile past K; lands in adjacent ws region, never consumed)
        stage_issue(lA0, lB0, gA, gB); gA += BK; gB += BK;
        asm volatile("s_wait_asynccnt 0x4" ::: "memory"); // previous tile (buf1) landed
        __syncthreads();
        compute_tile(As[1], Bs[1], acc, wmi, wni, lr, lh);
    }

    // epilogue: C/D layout — lane: n = lane%16; VGPR r -> m = r + (lane>=16 ? 8 : 0)
    #pragma unroll
    for (int mt = 0; mt < 4; ++mt) {
        #pragma unroll
        for (int nt = 0; nt < 2; ++nt) {
            int n = n0 + wni * 32 + nt * 16 + lr;
            float bv = bias[n];
            #pragma unroll
            for (int r = 0; r < 8; ++r) {
                int m = m0 + wmi * 64 + mt * 16 + lh * 8 + r;
                float val = acc[mt][nt][r] + bv;
                if (relu) val = fmaxf(val, 0.0f);
                size_t off = (size_t)m * DD + n;
                if (storef32) outf[off] = val;
                else          outb[off] = f2bf(val);
            }
        }
    }
}

extern "C" void kernel_launch(void* const* d_in, const int* in_sizes, int n_in,
                              void* d_out, int out_size, void* d_ws, size_t ws_size,
                              hipStream_t stream) {
    char* ws = (char*)d_ws;
    unsigned short* act0 = (unsigned short*)(ws);                                  // 32MB
    unsigned short* act1 = (unsigned short*)(ws + (size_t)NW * 2);                 // 32MB
    unsigned short* wmbf = (unsigned short*)(ws + (size_t)NW * 4);                 // 32MB
    float*    biasbuf = (float*)   (ws + (size_t)NW * 6);                          // 16KB
    unsigned* hist    = (unsigned*)(ws + (size_t)NW * 6 + DD * sizeof(float));     // 1KB
    uint2*    sel     = (uint2*)   (ws + (size_t)NW * 6 + DD * sizeof(float) + 256 * sizeof(unsigned));

    const float* x = (const float*)d_in[0];
    k_convert<<<4096, 256, 0, stream>>>(x, act0, NW);

    unsigned short* cur_in = act0;
    unsigned short* cur_out = act1;
    const unsigned shifts[4] = {24u, 16u, 8u, 0u};

    for (int l = 0; l < 4; ++l) {
        const float*    W  = (const float*)   d_in[1 + 4 * l];
        const float*    b  = (const float*)   d_in[2 + 4 * l];
        const unsigned* M  = (const unsigned*)d_in[3 + 4 * l];
        const float*    mb = (const float*)   d_in[4 + 4 * l];

        // exact k-th-smallest threshold on M's IEEE bit pattern (radix select, 4 byte-rounds)
        k_sel_init<<<1, 256, 0, stream>>>(hist, sel, (unsigned)KW);
        for (int rdx = 0; rdx < 4; ++rdx) {
            k_hist<<<2048, 256, 0, stream>>>(M, NW, shifts[rdx], sel, hist);
            k_scan<<<1, 256, 0, stream>>>(hist, sel);
        }
        k_mask_w<<<4096, 256, 0, stream>>>(W, M, sel, wmbf, NW);
        k_mask_bias<<<1, 1024, 0, stream>>>(b, mb, biasbuf);

        int last = (l == 3);
        k_gemm<<<dim3(DD / BN, DD / BM), 256, 0, stream>>>(
            cur_in, wmbf, biasbuf,
            last ? (unsigned short*)nullptr : cur_out,
            last ? (float*)d_out : (float*)nullptr,
            last ? 0 : 1, last);

        unsigned short* t = cur_in; cur_in = cur_out; cur_out = t;
    }
}
